// DefGoalNet_8100308320379
// MI455X (gfx1250) — matrix operand order repaired and checked by reading, not compiled, not run on hardware
//
// DefGoalNet forward for MI455X (gfx1250, wave32, WMMA).
//
// Precision/data-path choices (reasoned from MI455X specs, compile-only):
//  * density (4x4096x4096 pairwise + exp) -> VALU + v_exp_f32 with xyz in LDS
//    (48KB of 320KB/WGP); inner dim is 3 so matrix units don't apply.
//  * every GEMM-shaped op (bcks,bwks->bscw einsum, SA linear, fc1/2/3) ->
//    v_wmma_f32_16x16x32_f16 (f32 accumulate), K always a multiple of 32.
//    One wave (32 threads) per 16x16 output tile; EXEC all-ones at the WMMA.
//    Operand vectors built with constant-index initializers so they live in
//    VGPRs (no scratch round-trip).
//  * all intermediates (~45MB) << 192MB L2, so HBM BW is irrelevant here.
//
// d_in order assumed: tissue_pc, kidney_pc, then params leaves in JAX
// sorted-dict order (fc1_W,fc1_b,fc2_W,fc2_b,fc3_W,fc3_b,gn1_b,gn1_g,gn2_b,
// gn2_g, sa1, sa1_k, sa2, sa2_k, sa3, sa3_k; each sa: bn_b,bn_g,dn[3]{W,b,be,g},
// lin_W,lin_b,mlp{W,b,be,g},wn[3]{W,b,be,g}).  Requires ws_size >= ~64MB.

#include <hip/hip_runtime.h>
#include <cstdint>
#include <cstddef>

#define BATCH 4
#define NPTS  4096

typedef __attribute__((ext_vector_type(16))) _Float16 v16h;
typedef __attribute__((ext_vector_type(8)))  float    v8f;

// ---------------- WMMA lane-layout helpers (ISA 7.12.2, 16-bit operands) ----
__device__ __forceinline__ int a_koff(int lane, int j){
  // A 16x32: lanes0-15 hold K 0-7 (j<8) and 16-23 (j>=8); lanes16-31: 8-15 / 24-31
  int base = (j < 8) ? ((lane < 16) ? 0 : 8) : ((lane < 16) ? 16 : 24);
  return base + (j & 7);
}
__device__ __forceinline__ int b_koff(int lane, int j){
  // B 32x16: lanes0-15 hold K 0-7 (j<8) and 16-23 (j>=8); lanes16-31: 8-15 / 24-31
  return a_koff(lane, j);
}

// ---------------- density: inv_d[b,n] = 2.5*bw*N / sum_m exp(-d2/(2 bw^2)) ---
__global__ void k_density(const float* __restrict__ xyzT, float* __restrict__ inv_d,
                          int N, float bw){
  __shared__ float sx[4096], sy[4096], sz[4096];
  int b = blockIdx.y;
  const float* px = xyzT + (size_t)b * 3 * N;
  for(int i = threadIdx.x; i < N; i += blockDim.x){
    sx[i] = px[i]; sy[i] = px[N + i]; sz[i] = px[2 * N + i];
  }
  __syncthreads();
  int n = blockIdx.x * blockDim.x + threadIdx.x;
  if(n >= N) return;
  float xn = sx[n], yn = sy[n], zn = sz[n];
  float c = -1.0f / (2.0f * bw * bw);
  float acc = 0.f;
  for(int m = 0; m < N; ++m){
    float dx = xn - sx[m], dy = yn - sy[m], dz = zn - sz[m];
    acc += __expf((dx*dx + dy*dy + dz*dz) * c);   // v_exp_f32 on TRANS pipe
  }
  inv_d[(size_t)b * N + n] = (2.5f * bw * (float)N) / acc;
}

// ---------------- farthest point sampling (sequential, 1 WG per batch) ------
__global__ void k_fps(const float* __restrict__ xyzT, int* __restrict__ fidx,
                      int N, int npoint){
  int b = blockIdx.x, tid = threadIdx.x;
  __shared__ float dist[4096];
  __shared__ float rv[256]; __shared__ int ri[256]; __shared__ int sfar;
  for(int i = tid; i < N; i += 256) dist[i] = 1e10f;
  if(tid == 0) sfar = 0;
  __syncthreads();
  const float* px = xyzT + (size_t)b * 3 * N;
  for(int it = 0; it < npoint; ++it){
    int far = sfar;
    if(tid == 0) fidx[b * npoint + it] = far;
    float cx = px[far], cy = px[N + far], cz = px[2 * N + far];
    float bv = -1e30f; int bi = 0;
    for(int i = tid; i < N; i += 256){
      float dx = px[i] - cx, dy = px[N + i] - cy, dz = px[2 * N + i] - cz;
      float nd = fminf(dist[i], dx*dx + dy*dy + dz*dz);
      dist[i] = nd;
      if(nd > bv){ bv = nd; bi = i; }
    }
    rv[tid] = bv; ri[tid] = bi;
    __syncthreads();
    for(int s = 128; s > 0; s >>= 1){
      if(tid < s){
        bool take = (rv[tid+s] > rv[tid]) || (rv[tid+s] == rv[tid] && ri[tid+s] < ri[tid]);
        if(take){ rv[tid] = rv[tid+s]; ri[tid] = ri[tid+s]; }
      }
      __syncthreads();
    }
    if(tid == 0) sfar = ri[0];
    __syncthreads();
  }
}

// ---------------- gather new_xyz (B,S,3) + transposed (B,3,S) ---------------
__global__ void k_gather_xyz(const float* __restrict__ xyzT, const int* __restrict__ fidx,
                             float* __restrict__ new_xyz, float* __restrict__ xyzT_out,
                             int N, int S){
  int t = blockIdx.x * blockDim.x + threadIdx.x;
  if(t >= BATCH * S) return;
  int b = t / S, s = t % S;
  int j = fidx[t];
  for(int d = 0; d < 3; ++d){
    float v = xyzT[((size_t)b * 3 + d) * N + j];
    new_xyz[(size_t)t * 3 + d] = v;
    xyzT_out[((size_t)b * 3 + d) * S + s] = v;
  }
}

// ---------------- kNN: K smallest distances, ascending (1 WG per query) -----
__global__ void k_knn(const float* __restrict__ xyzT, const float* __restrict__ new_xyz,
                      int* __restrict__ idx, int N, int S, int K){
  int bs = blockIdx.x, b = bs / S, tid = threadIdx.x;
  __shared__ float dist[4096];
  __shared__ float rv[256]; __shared__ int ri[256];
  float qx = new_xyz[(size_t)bs*3], qy = new_xyz[(size_t)bs*3+1], qz = new_xyz[(size_t)bs*3+2];
  const float* px = xyzT + (size_t)b * 3 * N;
  for(int i = tid; i < N; i += 256){
    float dx = px[i]-qx, dy = px[N+i]-qy, dz = px[2*N+i]-qz;
    dist[i] = dx*dx + dy*dy + dz*dz;
  }
  __syncthreads();
  for(int j = 0; j < K; ++j){
    float bv = 3.2e38f; int bi = 0;
    for(int i = tid; i < N; i += 256){
      float v = dist[i];
      if(v < bv){ bv = v; bi = i; }
    }
    rv[tid] = bv; ri[tid] = bi;
    __syncthreads();
    for(int s = 128; s > 0; s >>= 1){
      if(tid < s){
        bool take = (rv[tid+s] < rv[tid]) || (rv[tid+s] == rv[tid] && ri[tid+s] < ri[tid]);
        if(take){ rv[tid] = rv[tid+s]; ri[tid] = ri[tid+s]; }
      }
      __syncthreads();
    }
    if(tid == 0){ idx[(size_t)bs * K + j] = ri[0]; dist[ri[0]] = 3.3e38f; }
    __syncthreads();
  }
}

// ---------------- grouping: xin(B,3+C,K,S), win(B,3,K,S), dsin(B,1,K,S) -----
__global__ void k_group(const float* __restrict__ xyzT, const float* __restrict__ pts,
                        const float* __restrict__ inv_d, const int* __restrict__ idx,
                        const float* __restrict__ new_xyz,
                        float* __restrict__ xin, float* __restrict__ win,
                        float* __restrict__ dsin, int N, int S, int K, int C){
  int bs = blockIdx.x, b = bs / S, s = bs % S, k = threadIdx.x;
  __shared__ float sd[64]; __shared__ float smax;
  int j = idx[(size_t)bs * K + k];
  int Ci = 3 + C;
  size_t xbase = (size_t)b * Ci * K * S;
  for(int d = 0; d < 3; ++d){
    float v = xyzT[((size_t)b*3 + d) * N + j] - new_xyz[(size_t)bs*3 + d];
    xin[xbase + ((size_t)d * K + k) * S + s] = v;
    win[((size_t)(b*3 + d) * K + k) * S + s] = v;
  }
  for(int c = 0; c < C; ++c)
    xin[xbase + ((size_t)(3 + c) * K + k) * S + s] = pts[((size_t)b*C + c) * N + j];
  float dv = inv_d[(size_t)b * N + j];
  sd[k] = dv;
  __syncthreads();
  if(k == 0){ float m = sd[0]; for(int t = 1; t < K; ++t) m = fmaxf(m, sd[t]); smax = m; }
  __syncthreads();
  dsin[((size_t)b * K + k) * S + s] = dv / smax;
}

// ---------------- group_all (sa3): K=N, S=1 ---------------------------------
__global__ void k_group_all(const float* __restrict__ xyzT, const float* __restrict__ pts,
                            const float* __restrict__ inv_d,
                            float* __restrict__ xin, float* __restrict__ win,
                            float* __restrict__ dsin, int N, int C){
  int b = blockIdx.x, tid = threadIdx.x;
  __shared__ float red[128]; __shared__ float smax;
  float mx = -1e30f;
  for(int i = tid; i < N; i += blockDim.x) mx = fmaxf(mx, inv_d[(size_t)b*N + i]);
  red[tid] = mx;
  __syncthreads();
  for(int s = 64; s > 0; s >>= 1){
    if(tid < s) red[tid] = fmaxf(red[tid], red[tid+s]);
    __syncthreads();
  }
  if(tid == 0) smax = red[0];
  __syncthreads();
  int Ci = 3 + C;
  for(int k = tid; k < N; k += blockDim.x){
    for(int d = 0; d < 3; ++d){
      float v = xyzT[((size_t)b*3 + d) * N + k];
      xin[((size_t)b*Ci + d) * N + k] = v;
      win[((size_t)b*3  + d) * N + k] = v;
    }
    for(int c = 0; c < C; ++c)
      xin[((size_t)b*Ci + 3 + c) * N + k] = pts[((size_t)b*C + c) * N + k];
    dsin[(size_t)b * N + k] = inv_d[(size_t)b*N + k] / smax;
  }
}

// ---------------- 1x1 conv: y[b,o,ks] = sum_c W[o,c] x[b,c,ks] + bias[o] ----
__global__ void k_conv(const float* __restrict__ x, const float* __restrict__ W,
                       const float* __restrict__ bias, float* __restrict__ y,
                       int Ci, int Co, int KS){
  size_t t = (size_t)blockIdx.x * blockDim.x + threadIdx.x;
  size_t total = (size_t)BATCH * Co * KS;
  if(t >= total) return;
  int ks = (int)(t % KS); size_t r = t / KS; int o = (int)(r % Co); int b = (int)(r / Co);
  const float* xb = x + (size_t)b * Ci * KS + ks;
  const float* w  = W + (size_t)o * Ci;
  float acc = bias[o];
  for(int c = 0; c < Ci; ++c) acc += w[c] * xb[(size_t)c * KS];
  y[t] = acc;
}

// ---------------- fused batch-wide BN + ReLU (one WG per channel) -----------
__global__ void k_bn_relu(float* __restrict__ y, const float* __restrict__ g,
                          const float* __restrict__ be, int Co, int KS){
  int o = blockIdx.x, tid = threadIdx.x;
  __shared__ float r1[256], r2[256];
  float s = 0.f, s2 = 0.f;
  for(int b = 0; b < BATCH; ++b){
    float* p = y + ((size_t)b * Co + o) * KS;
    for(int i = tid; i < KS; i += 256){ float v = p[i]; s += v; s2 += v*v; }
  }
  r1[tid] = s; r2[tid] = s2;
  __syncthreads();
  for(int st = 128; st > 0; st >>= 1){
    if(tid < st){ r1[tid] += r1[tid+st]; r2[tid] += r2[tid+st]; }
    __syncthreads();
  }
  float cnt = (float)BATCH * (float)KS;
  float m = r1[0] / cnt, var = r2[0] / cnt - m * m;
  float sc = g[o] * rsqrtf(var + 1e-5f), sh = be[o] - m * sc;
  for(int b = 0; b < BATCH; ++b){
    float* p = y + ((size_t)b * Co + o) * KS;
    for(int i = tid; i < KS; i += 256) p[i] = fmaxf(p[i] * sc + sh, 0.f);
  }
}

// ---------------- x *= dscale -----------------------------------------------
__global__ void k_mul(float* __restrict__ x, const float* __restrict__ d, int Co, int KS){
  size_t t = (size_t)blockIdx.x * blockDim.x + threadIdx.x;
  size_t total = (size_t)BATCH * Co * KS;
  if(t >= total) return;
  int ks = (int)(t % KS); int b = (int)(t / ((size_t)Co * KS));
  x[t] *= d[(size_t)b * KS + ks];
}

// ---------------- einsum bcks,bwks->bscw via WMMA (1 wave / 16x16 tile) -----
// x:(B,Co,Kd,S)  w:(B,16,Kd,S)  out:(B,S,Co*16) with feature index c*16+w
__global__ void k_einsum_wmma(const float* __restrict__ x, const float* __restrict__ w,
                              float* __restrict__ out, int Co, int Kd, int S){
  int blk = blockIdx.x;
  int nct = Co >> 4;
  int ct = blk % nct; int bs = blk / nct; int b = bs / S, s = bs % S;
  int l = threadIdx.x;
  int m16 = l & 15;
  const float* xrow = x + ((size_t)b * Co + (ct*16 + m16)) * Kd * S + s; // k stride = S
  const float* wrow = w + ((size_t)b * 16 + m16) * Kd * S + s;           // k stride = S
  v8f acc = {};
  for(int kc = 0; kc < Kd; kc += 32){
    const float* xa = xrow + (size_t)kc * S;
    const float* wb = wrow + (size_t)kc * S;
    // Constant-index initializers -> operands assembled in VGPRs (no scratch).
#define EA(j) (_Float16)xa[(size_t)a_koff(l, (j)) * S]
#define EB(j) (_Float16)wb[(size_t)b_koff(l, (j)) * S]
    v16h av = { EA(0), EA(1), EA(2),  EA(3),  EA(4),  EA(5),  EA(6),  EA(7),
                EA(8), EA(9), EA(10), EA(11), EA(12), EA(13), EA(14), EA(15) };
    v16h bv = { EB(0), EB(1), EB(2),  EB(3),  EB(4),  EB(5),  EB(6),  EB(7),
                EB(8), EB(9), EB(10), EB(11), EB(12), EB(13), EB(14), EB(15) };
#undef EA
#undef EB
    acc = __builtin_amdgcn_wmma_f32_16x16x32_f16(false, av, false, bv,
                                                 (short)0, acc, false, false);
  }
  int rbase = (l < 16) ? 0 : 8;
  for(int r = 0; r < 8; ++r){
    int c = ct * 16 + r + rbase;
    out[((size_t)(b * S + s) * Co + c) * 16 + m16] = acc[r];
  }
}

// ---------------- generic linear via WMMA: C[M,N] = A[M,K] * Wt[N,K]^T + b --
__global__ void k_linear_wmma(const float* __restrict__ A, const float* __restrict__ Wt,
                              const float* __restrict__ bias, float* __restrict__ C,
                              int M, int N, int K){
  int blk = blockIdx.x;
  int ntile = N >> 4;
  int nt = blk % ntile, mt = blk / ntile;
  int l = threadIdx.x;
  int m16 = l & 15;
  int row = mt * 16 + m16;
  bool rowok = row < M;
  const float* arow = A  + (size_t)(rowok ? row : 0) * K;  // clamped, zeroed below
  const float* wrow = Wt + (size_t)(nt * 16 + m16) * K;
  const v16h vzero = {};
  v8f acc = {};
  for(int kc = 0; kc < K; kc += 32){
    if(kc + 32 < K)
      __builtin_prefetch(&wrow[kc + 32], 0, 3);            // global_prefetch_b8
#define EA(j) (_Float16)arow[kc + a_koff(l, (j))]
#define EB(j) (_Float16)wrow[kc + b_koff(l, (j))]
    v16h av = { EA(0), EA(1), EA(2),  EA(3),  EA(4),  EA(5),  EA(6),  EA(7),
                EA(8), EA(9), EA(10), EA(11), EA(12), EA(13), EA(14), EA(15) };
    v16h bv = { EB(0), EB(1), EB(2),  EB(3),  EB(4),  EB(5),  EB(6),  EB(7),
                EB(8), EB(9), EB(10), EB(11), EB(12), EB(13), EB(14), EB(15) };
#undef EA
#undef EB
    if(!rowok) av = vzero;   // padding rows contribute zero
    acc = __builtin_amdgcn_wmma_f32_16x16x32_f16(false, av, false, bv,
                                                 (short)0, acc, false, false);
  }
  int rbase = (l < 16) ? 0 : 8;
  int col = nt * 16 + m16;
  for(int r = 0; r < 8; ++r){
    int rr = mt * 16 + r + rbase;
    if(rr < M) C[(size_t)rr * N + col] = acc[r] + bias[col];
  }
}

// ---------------- BN1d over (b,s) + ReLU + transpose to (B,Co,S) ------------
__global__ void k_bn1d_relu_t(const float* __restrict__ in, float* __restrict__ out,
                              const float* __restrict__ g, const float* __restrict__ be,
                              int S, int Co){
  int o = blockIdx.x, tid = threadIdx.x;
  int M = BATCH * S;
  __shared__ float r1[256], r2[256];
  float s = 0.f, s2 = 0.f;
  for(int r = tid; r < M; r += 256){ float v = in[(size_t)r * Co + o]; s += v; s2 += v*v; }
  r1[tid] = s; r2[tid] = s2;
  __syncthreads();
  for(int st = 128; st > 0; st >>= 1){
    if(tid < st){ r1[tid] += r1[tid+st]; r2[tid] += r2[tid+st]; }
    __syncthreads();
  }
  float m = r1[0] / (float)M, var = r2[0] / (float)M - m * m;
  float sc = g[o] * rsqrtf(var + 1e-5f), sh = be[o] - m * sc;
  for(int r = tid; r < M; r += 256){
    int b = r / S, ss = r % S;
    float v = in[(size_t)r * Co + o] * sc + sh;
    out[((size_t)b * Co + o) * S + ss] = fmaxf(v, 0.f);
  }
}

// ---------------- concat two (B,256) -> (B,512) -----------------------------
__global__ void k_concat(const float* __restrict__ a, const float* __restrict__ b,
                         float* __restrict__ o){
  int t = blockIdx.x * blockDim.x + threadIdx.x;
  if(t >= BATCH * 512) return;
  int bb = t / 512, f = t % 512;
  o[t] = (f < 256) ? a[bb * 256 + f] : b[bb * 256 + f - 256];
}

// ---------------- groupnorm over features + ReLU (per sample) ---------------
__global__ void k_gn_relu(float* __restrict__ x, const float* __restrict__ g,
                          const float* __restrict__ bsh, int F){
  int b = blockIdx.x, tid = threadIdx.x;
  __shared__ float r1[256], r2[256];
  float s = 0.f, s2 = 0.f;
  for(int i = tid; i < F; i += 256){ float v = x[(size_t)b*F + i]; s += v; s2 += v*v; }
  r1[tid] = s; r2[tid] = s2;
  __syncthreads();
  for(int st = 128; st > 0; st >>= 1){
    if(tid < st){ r1[tid] += r1[tid+st]; r2[tid] += r2[tid+st]; }
    __syncthreads();
  }
  float m = r1[0] / (float)F, var = r2[0] / (float)F - m * m;
  float rs = rsqrtf(var + 1e-5f);
  for(int i = tid; i < F; i += 256){
    float v = (x[(size_t)b*F + i] - m) * rs * g[i] + bsh[i];
    x[(size_t)b*F + i] = fmaxf(v, 0.f);
  }
}

// ======================= host side ==========================================
namespace {

struct Bump {
  char* base; size_t off; size_t cap;
  float* f(size_t n){ size_t o = (off + 255) & ~(size_t)255; off = o + n * 4; return (float*)(base + o); }
  int*   i(size_t n){ size_t o = (off + 255) & ~(size_t)255; off = o + n * 4; return (int*)(base + o); }
};

struct ConvP { const float *W, *b, *be, *g; };
struct SaP   { const float *bn_b, *bn_g; ConvP dn[3]; const float *lin_W, *lin_b; ConvP mlp; ConvP wn[3]; };

struct Walker {
  void* const* d; int i;
  const float* nf(){ return (const float*)d[i++]; }
  ConvP conv(){ ConvP c; c.W = nf(); c.b = nf(); c.be = nf(); c.g = nf(); return c; }
  SaP sa(){
    SaP s;
    s.bn_b = nf(); s.bn_g = nf();
    for(int k = 0; k < 3; ++k) s.dn[k] = conv();
    s.lin_W = nf(); s.lin_b = nf();
    s.mlp = conv();
    for(int k = 0; k < 3; ++k) s.wn[k] = conv();
    return s;
  }
};

void conv_bn_relu(const float* x, float* y, const ConvP& c, int Ci, int Co, int KS,
                  hipStream_t st){
  size_t total = (size_t)BATCH * Co * KS;
  int blocks = (int)((total + 255) / 256);
  k_conv<<<blocks, 256, 0, st>>>(x, c.W, c.b, y, Ci, Co, KS);
  k_bn_relu<<<Co, 256, 0, st>>>(y, c.g, c.be, Co, KS);
}

// One SA layer. xyzT:(B,3,N), pts:(B,C,N). Outputs xyzT_out:(B,3,S) (if !group_all)
// and pout:(B,Co,S). Bump passed by value: arena resets per layer.
void run_sa(const float* xyzT, const float* pts, int N, int C, int S, int K, int Co,
            float bw, bool group_all, const SaP& P,
            float* xyzT_out, float* pout, Bump A, hipStream_t st){
  int Ci = 3 + C;
  float* inv_d = A.f((size_t)BATCH * N);
  k_density<<<dim3((N + 255) / 256, BATCH), 256, 0, st>>>(xyzT, inv_d, N, bw);

  int Kd = group_all ? N : K;
  int Sd = group_all ? 1 : S;
  int KS = Kd * Sd;
  float* xin  = A.f((size_t)BATCH * Ci * KS);
  float* win  = A.f((size_t)BATCH * 3  * KS);
  float* dsin = A.f((size_t)BATCH * KS);

  if(!group_all){
    int* fidx = A.i((size_t)BATCH * S);
    k_fps<<<BATCH, 256, 0, st>>>(xyzT, fidx, N, S);
    float* nx = A.f((size_t)BATCH * S * 3);
    k_gather_xyz<<<(BATCH * S + 255) / 256, 256, 0, st>>>(xyzT, fidx, nx, xyzT_out, N, S);
    int* idx = A.i((size_t)BATCH * S * K);
    k_knn<<<BATCH * S, 256, 0, st>>>(xyzT, nx, idx, N, S, K);
    k_group<<<BATCH * S, K, 0, st>>>(xyzT, pts, inv_d, idx, nx, xin, win, dsin, N, S, K, C);
  } else {
    k_group_all<<<BATCH, 128, 0, st>>>(xyzT, pts, inv_d, xin, win, dsin, N, C);
  }

  // mlp: Ci -> Co
  float* x = A.f((size_t)BATCH * Co * KS);
  conv_bn_relu(xin, x, P.mlp, Ci, Co, KS, st);

  // density net: 1 -> 16 -> 8 -> 1
  float* d1 = A.f((size_t)BATCH * 16 * KS); conv_bn_relu(dsin, d1, P.dn[0], 1, 16, KS, st);
  float* d2 = A.f((size_t)BATCH * 8  * KS); conv_bn_relu(d1,   d2, P.dn[1], 16, 8, KS, st);
  float* d3 = A.f((size_t)BATCH * 1  * KS); conv_bn_relu(d2,   d3, P.dn[2], 8,  1, KS, st);
  {
    size_t total = (size_t)BATCH * Co * KS;
    k_mul<<<(int)((total + 255) / 256), 256, 0, st>>>(x, d3, Co, KS);
  }

  // weight net: 3 -> 8 -> 8 -> 16
  float* w1 = A.f((size_t)BATCH * 8  * KS); conv_bn_relu(win, w1, P.wn[0], 3, 8,  KS, st);
  float* w2 = A.f((size_t)BATCH * 8  * KS); conv_bn_relu(w1,  w2, P.wn[1], 8, 8,  KS, st);
  float* w3 = A.f((size_t)BATCH * 16 * KS); conv_bn_relu(w2,  w3, P.wn[2], 8, 16, KS, st);

  // einsum bcks,bwks->bscw  (WMMA f16)
  float* eout = A.f((size_t)BATCH * Sd * Co * 16);
  k_einsum_wmma<<<BATCH * Sd * (Co / 16), 32, 0, st>>>(x, w3, eout, Co, Kd, Sd);

  // linear: (B*Sd, 16*Co) x (Co, 16*Co)^T + lin_b  (WMMA f16)
  float* lout = A.f((size_t)BATCH * Sd * Co);
  int M = BATCH * Sd, Kk = 16 * Co, Nn = Co;
  k_linear_wmma<<<((M + 15) / 16) * (Nn / 16), 32, 0, st>>>(eout, P.lin_W, P.lin_b,
                                                            lout, M, Nn, Kk);
  // BN over (b,s) + ReLU, transposed output (B,Co,Sd)
  k_bn1d_relu_t<<<Co, 256, 0, st>>>(lout, pout, P.bn_g, P.bn_b, Sd, Co);
}

void run_branch(const float* pc, const SaP& s1, const SaP& s2, const SaP& s3,
                float* emb, Bump A, hipStream_t st){
  // carry buffers survive across SA layers of this branch
  float* xyzT1 = A.f((size_t)BATCH * 3 * 512);
  float* p1    = A.f((size_t)BATCH * 64 * 512);
  float* xyzT2 = A.f((size_t)BATCH * 3 * 128);
  float* p2    = A.f((size_t)BATCH * 128 * 128);
  run_sa(pc,    pc, NPTS, 3,   512, 32, 64,  0.1f, false, s1, xyzT1, p1, A, st);
  run_sa(xyzT1, p1, 512,  64,  128, 64, 128, 0.2f, false, s2, xyzT2, p2, A, st);
  run_sa(xyzT2, p2, 128,  128, 1,   128, 256, 0.4f, true, s3, nullptr, emb, A, st);
}

} // namespace

extern "C" void kernel_launch(void* const* d_in, const int* in_sizes, int n_in,
                              void* d_out, int out_size, void* d_ws, size_t ws_size,
                              hipStream_t stream){
  (void)in_sizes; (void)n_in; (void)out_size;
  Walker W{d_in, 0};
  const float* tissue = W.nf();
  const float* kidney = W.nf();
  const float* fc1_W = W.nf(); const float* fc1_b = W.nf();
  const float* fc2_W = W.nf(); const float* fc2_b = W.nf();
  const float* fc3_W = W.nf(); const float* fc3_b = W.nf();
  const float* gn1_b = W.nf(); const float* gn1_g = W.nf();
  const float* gn2_b = W.nf(); const float* gn2_g = W.nf();
  SaP sa1  = W.sa(); SaP sa1k = W.sa();
  SaP sa2  = W.sa(); SaP sa2k = W.sa();
  SaP sa3  = W.sa(); SaP sa3k = W.sa();

  Bump A{(char*)d_ws, 0, ws_size};
  float* emb1 = A.f((size_t)BATCH * 256);
  float* emb2 = A.f((size_t)BATCH * 256);
  float* xcat = A.f((size_t)BATCH * 512);
  float* h1   = A.f((size_t)BATCH * 256);
  float* h2   = A.f((size_t)BATCH * 256);

  run_branch(tissue, sa1,  sa2,  sa3,  emb1, A, stream);
  run_branch(kidney, sa1k, sa2k, sa3k, emb2, A, stream);

  k_concat<<<(BATCH * 512 + 255) / 256, 256, 0, stream>>>(emb1, emb2, xcat);
  // head: fc1 (512->256) + GN + ReLU, fc2 (256->256) + GN + ReLU, fc3 (256->12288)
  k_linear_wmma<<<1 * (256 / 16), 32, 0, stream>>>(xcat, fc1_W, fc1_b, h1, BATCH, 256, 512);
  k_gn_relu<<<BATCH, 256, 0, stream>>>(h1, gn1_g, gn1_b, 256);
  k_linear_wmma<<<1 * (256 / 16), 32, 0, stream>>>(h1, fc2_W, fc2_b, h2, BATCH, 256, 256);
  k_gn_relu<<<BATCH, 256, 0, stream>>>(h2, gn2_g, gn2_b, 256);
  // fc3 writes the final (B,3*NPTS) = d_out directly (reshape is a no-op)
  k_linear_wmma<<<1 * (12288 / 16), 32, 0, stream>>>(h2, fc3_W, fc3_b, (float*)d_out,
                                                     BATCH, 12288, 256);
}
